// InstanceDenseUNet_70884140253246
// MI455X (gfx1250) — compile-verified, matrix operand order checked by reading, not run on hardware
//
#include <hip/hip_runtime.h>
#include <hip/hip_bf16.h>

typedef __attribute__((ext_vector_type(16))) __bf16 v16bf;
typedef __attribute__((ext_vector_type(8)))  __bf16 v8bf;
typedef __attribute__((ext_vector_type(8)))  float  v8f;

#define TPB 256

// ---------------------------------------------------------------- utilities
__global__ void zero_f32_k(float* __restrict__ p, long long n) {
  long long t = (long long)blockIdx.x * blockDim.x + threadIdx.x;
  if (t < n) p[t] = 0.f;
}

// feats (N x C f32) -> bf16 (N+1) x Cp, zero-padding extra cols and pad row
__global__ void convert_pad_k(const float* __restrict__ x, __bf16* __restrict__ out,
                              int N, int C, int Cp) {
  long long t = (long long)blockIdx.x * blockDim.x + threadIdx.x;
  long long total = (long long)(N + 1) * Cp;
  if (t >= total) return;
  int i = (int)(t / Cp), c = (int)(t % Cp);
  float v = (i < N && c < C) ? x[(size_t)i * C + c] : 0.f;
  out[t] = (__bf16)v;
}

// per-channel mean / inv-std (population var, eps=1e-4) — one block per channel
__global__ void bn_stats_k(const float* __restrict__ x, float* __restrict__ mean,
                           float* __restrict__ invstd, int N, int C) {
  int c = blockIdx.x;
  float s = 0.f, s2 = 0.f;
  for (int i = threadIdx.x; i < N; i += blockDim.x) {
    float v = x[(size_t)i * C + c];
    s += v; s2 += v * v;
  }
  __shared__ float sh0[TPB], sh1[TPB];
  sh0[threadIdx.x] = s; sh1[threadIdx.x] = s2;
  __syncthreads();
  for (int o = TPB / 2; o > 0; o >>= 1) {
    if ((int)threadIdx.x < o) {
      sh0[threadIdx.x] += sh0[threadIdx.x + o];
      sh1[threadIdx.x] += sh1[threadIdx.x + o];
    }
    __syncthreads();
  }
  if (threadIdx.x == 0) {
    float m = sh0[0] / (float)N;
    float var = sh1[0] / (float)N - m * m;
    mean[c]   = m;
    invstd[c] = rsqrtf(var + 1e-4f);
  }
}

// BN + ReLU, write bf16 into a (N+1) x Cp buffer at column offset colOff (pad row zeroed)
__global__ void bnrelu_bf16_k(const float* __restrict__ x, const float* __restrict__ mean,
                              const float* __restrict__ invstd, __bf16* __restrict__ out,
                              int N, int C, int Cp, int colOff) {
  long long t = (long long)blockIdx.x * blockDim.x + threadIdx.x;
  long long total = (long long)(N + 1) * C;
  if (t >= total) return;
  int i = (int)(t / C), c = (int)(t % C);
  float v = 0.f;
  if (i < N) {
    v = (x[(size_t)i * C + c] - mean[c]) * invstd[c];
    v = fmaxf(v, 0.f);
  }
  out[(size_t)i * Cp + colOff + c] = (__bf16)v;
}

// BN + ReLU to f32 (final feature)
__global__ void bnrelu_f32_k(const float* __restrict__ x, const float* __restrict__ mean,
                             const float* __restrict__ invstd, float* __restrict__ out,
                             int N, int C) {
  long long t = (long long)blockIdx.x * blockDim.x + threadIdx.x;
  long long total = (long long)N * C;
  if (t >= total) return;
  int i = (int)(t / C), c = (int)(t % C);
  float v = (x[(size_t)i * C + c] - mean[c]) * invstd[c];
  out[t] = fmaxf(v, 0.f);
}

// ------------------------------------------------- weight packing for WMMA B
// packed[((k*NT + nt)*KT + kt)*512 + lane*16 + e] ; 16x16x32 bf16 fragments.
// lane = hi*16 + ncol ; element e<8 -> K = hi*8+e ; e>=8 -> K = 16 + hi*8 + (e-8)
__global__ void pack_w_k(const float* __restrict__ W, __bf16* __restrict__ P,
                         int KK, int Cin, int Cout, int Cinp) {
  int NT = Cout >> 4, KT = Cinp >> 5;
  long long total = (long long)KK * NT * KT * 512;
  long long t = (long long)blockIdx.x * blockDim.x + threadIdx.x;
  if (t >= total) return;
  int e    = (int)(t & 15);
  int lane = (int)((t >> 4) & 31);
  long long rest = t >> 9;
  int kt = (int)(rest % KT); rest /= KT;
  int nt = (int)(rest % NT);
  int k  = (int)(rest / NT);
  int hi = lane >> 4, ncol = lane & 15;
  int kk = (e < 8) ? (hi * 8 + e) : (16 + hi * 8 + (e - 8));
  int K = kt * 32 + kk, col = nt * 16 + ncol;
  float val = (K < Cin) ? W[((size_t)k * Cin + K) * Cout + col] : 0.f;
  P[t] = (__bf16)val;
}

// ---------------------------------------------- submanifold 3^3 conv (WMMA)
// one wave = one 16-row tile, owning ALL NT 16-col output tiles in registers:
// A fragment (gathered via rulebook) is loaded once per (k,kt) and reused NT times.
template <int NT>
__global__ __launch_bounds__(TPB) void subconv_wmma_k(
    const __bf16* __restrict__ xbf, const int* __restrict__ nbr,
    const __bf16* __restrict__ wp, float* __restrict__ out,
    int N, int Cp) {
  const int Cout = NT * 16;
  const int KT = Cp >> 5;
  const int rtiles = (N + 15) >> 4;
  int w = blockIdx.x * (blockDim.x >> 5) + ((int)threadIdx.x >> 5);
  if (w >= rtiles) return;            // wave-uniform exit: EXEC stays all-ones
  const int lane = (int)threadIdx.x & 31;
  const int r0 = w << 4;
  const int m  = lane & 15;
  const int hi = lane >> 4;
  const int row = r0 + m;
  const int rsafe = (row < N) ? row : (N - 1);

  v8f acc[NT];
#pragma unroll
  for (int nt = 0; nt < NT; ++nt) acc[nt] = (v8f){};

  const size_t wkstride = (size_t)NT * KT * 512;
  const __bf16* wk0 = wp + (size_t)lane * 16;

  for (int k = 0; k < 27; ++k) {
    int idx = (row < N) ? nbr[(size_t)rsafe * 27 + k] : N;   // N = zero pad row
    const __bf16* arow = xbf + (size_t)idx * Cp + hi * 8;
    const __bf16* wk   = wk0 + (size_t)k * wkstride;
    if (k + 1 < 27) __builtin_prefetch(wk0 + (size_t)(k + 1) * wkstride, 0, 0);
    for (int kt = 0; kt < KT; ++kt) {
      v8bf a0 = *(const v8bf*)(arow + kt * 32);        // K block {0..7}+sel
      v8bf a1 = *(const v8bf*)(arow + kt * 32 + 16);   // K block {16..23}+sel
      v16bf a = __builtin_shufflevector(a0, a1,
          0, 1, 2, 3, 4, 5, 6, 7, 8, 9, 10, 11, 12, 13, 14, 15);
#pragma unroll
      for (int nt = 0; nt < NT; ++nt) {
        v16bf b = *(const v16bf*)(wk + ((size_t)nt * KT + kt) * 512);
        acc[nt] = __builtin_amdgcn_wmma_f32_16x16x32_bf16(
            false, a, false, b, (short)0, acc[nt], false, false);
      }
    }
  }
  // D layout: lane<16 -> col=lane, rows r (VGPR r); lane>=16 -> col=lane-16, rows r+8
#pragma unroll
  for (int nt = 0; nt < NT; ++nt) {
    int col = nt * 16 + m;
#pragma unroll
    for (int r = 0; r < 8; ++r) {
      int orow = r0 + hi * 8 + r;
      if (orow < N) out[(size_t)orow * Cout + col] = acc[nt][r];
    }
  }
}

// -------------------------------- strided 2x down-conv: scatter-add per octant
__global__ void downconv_k(const __bf16* __restrict__ xbf, const int* __restrict__ parent,
                           const int* __restrict__ off, const float* __restrict__ W,
                           float* __restrict__ out, int N, int Cin, int Cp, int Cout) {
  long long t = (long long)blockIdx.x * blockDim.x + threadIdx.x;
  long long total = (long long)N * Cout;
  if (t >= total) return;
  int i = (int)(t / Cout), j = (int)(t % Cout);
  const float* Wk = W + (size_t)off[i] * Cin * Cout;
  const __bf16* xr = xbf + (size_t)i * Cp;
  float acc = 0.f;
  for (int c = 0; c < Cin; ++c)
    acc += (float)xr[c] * Wk[(size_t)c * Cout + j];
  atomicAdd(&out[(size_t)parent[i] * Cout + j], acc);
}

// -------------------------------- strided 2x deconv: broadcast parent feature
__global__ void deconv_k(const __bf16* __restrict__ xbf, const int* __restrict__ parent,
                         const int* __restrict__ off, const float* __restrict__ W,
                         float* __restrict__ out, int N, int Cin, int Cp, int Cout) {
  long long t = (long long)blockIdx.x * blockDim.x + threadIdx.x;
  long long total = (long long)N * Cout;
  if (t >= total) return;
  int i = (int)(t / Cout), j = (int)(t % Cout);
  const float* Wk = W + (size_t)off[i] * Cin * Cout;
  const __bf16* xr = xbf + (size_t)parent[i] * Cp;
  float acc = 0.f;
  for (int c = 0; c < Cin; ++c)
    acc += (float)xr[c] * Wk[(size_t)c * Cout + j];
  out[(size_t)i * Cout + j] = acc;
}

// -------------------------------- dense linear head (act: 0=none, 1=sigmoid)
__global__ void linear_k(const float* __restrict__ X, const float* __restrict__ W,
                         const float* __restrict__ b, float* __restrict__ Y,
                         int N, int Cin, int Cout, int act) {
  long long t = (long long)blockIdx.x * blockDim.x + threadIdx.x;
  long long total = (long long)N * Cout;
  if (t >= total) return;
  int i = (int)(t / Cout), j = (int)(t % Cout);
  float acc = b[j];
  for (int c = 0; c < Cin; ++c)
    acc += X[(size_t)i * Cin + c] * W[(size_t)c * Cout + j];
  if (act) acc = 1.f / (1.f + expf(-acc));
  Y[(size_t)i * Cout + j] = acc;
}

// ============================================================ host orchestration
static inline unsigned gdim(long long n, int tpb) {
  return (unsigned)((n + tpb - 1) / tpb);
}

extern "C" void kernel_launch(void* const* d_in, const int* in_sizes, int n_in,
                              void* d_out, int out_size, void* d_ws, size_t ws_size,
                              hipStream_t stream) {
  (void)n_in; (void)out_size; (void)ws_size;
  static const int P[7] = {32, 64, 96, 128, 160, 192, 224};

  // ---- input pointers (setup_inputs dict order, flattened) ----
  const float* feats = (const float*)d_in[0];
  const float* W_in  = (const float*)d_in[1];
  const float* W_pre[7];  for (int l = 0; l < 7; ++l) W_pre[l]  = (const float*)d_in[2 + l];
  const float* W_down[6]; for (int l = 0; l < 6; ++l) W_down[l] = (const float*)d_in[9 + l];
  const float* W_up[6];   for (int l = 0; l < 6; ++l) W_up[l]   = (const float*)d_in[15 + l];
  const float* W_post[6]; for (int l = 0; l < 6; ++l) W_post[l] = (const float*)d_in[21 + l];
  const float* Wc  = (const float*)d_in[27]; const float* bc  = (const float*)d_in[28];
  const float* Wfe = (const float*)d_in[29]; const float* bfe = (const float*)d_in[30];
  const float* Wle = (const float*)d_in[31]; const float* ble = (const float*)d_in[32];
  const float* Wfr = (const float*)d_in[33]; const float* bfr = (const float*)d_in[34];
  const float* Wlr = (const float*)d_in[35]; const float* blr = (const float*)d_in[36];
  const float* Wfd = (const float*)d_in[37]; const float* bfd = (const float*)d_in[38];
  const float* Wld = (const float*)d_in[39]; const float* bld = (const float*)d_in[40];
  const int* nbr[7];    for (int l = 0; l < 7; ++l) nbr[l]    = (const int*)d_in[41 + l];
  const int* parent[6]; for (int l = 0; l < 6; ++l) parent[l] = (const int*)d_in[48 + l];
  const int* offs[6];   for (int l = 0; l < 6; ++l) offs[l]   = (const int*)d_in[54 + l];

  int Nl[7];
  for (int l = 0; l < 7; ++l) Nl[l] = in_sizes[41 + l] / 27;
  const int N0 = Nl[0];

  // ---- bump allocator over workspace ----
  char* base = (char*)d_ws;
  size_t cur = 0;
  auto alloc = [&](size_t bytes) -> void* {
    void* p = (void*)(base + cur);
    cur = (cur + bytes + 255) & ~(size_t)255;
    return p;
  };

  size_t maxbf = (size_t)(N0 + 1) * 32;
  for (int l = 0; l < 6; ++l) {
    size_t s = (size_t)(Nl[l] + 1) * 2 * P[l];
    if (s > maxbf) maxbf = s;
  }
  { size_t s = (size_t)(Nl[6] + 1) * 224; if (s > maxbf) maxbf = s; }
  size_t maxact = 0, maxu = 0;
  for (int l = 0; l < 7; ++l) { size_t s = (size_t)Nl[l] * P[l]; if (s > maxact) maxact = s; }
  for (int l = 0; l < 6; ++l) { size_t s = (size_t)Nl[l] * P[l]; if (s > maxu) maxu = s; }
  auto packsz = [](int Cinp, int Cout) { return (size_t)27 * (Cout / 16) * (Cinp / 32) * 512; };
  size_t maxpack = packsz(32, 32);
  for (int l = 0; l < 7; ++l) { size_t s = packsz(P[l], P[l]); if (s > maxpack) maxpack = s; }
  for (int l = 0; l < 6; ++l) { size_t s = packsz(2 * P[l], P[l]); if (s > maxpack) maxpack = s; }

  float*  mean   = (float*)alloc(512 * sizeof(float));
  float*  invstd = (float*)alloc(512 * sizeof(float));
  __bf16* xbf    = (__bf16*)alloc(maxbf * sizeof(__bf16));
  float*  actA   = (float*)alloc(maxact * sizeof(float));
  float*  actB   = (float*)alloc(maxact * sizeof(float));
  float*  uBuf   = (float*)alloc(maxu * sizeof(float));
  __bf16* packed = (__bf16*)alloc(maxpack * sizeof(__bf16));
  float*  skip[6];
  for (int l = 0; l < 6; ++l) skip[l] = (float*)alloc((size_t)Nl[l] * P[l] * sizeof(float));

  // ---- launch helpers ----
  auto stats = [&](const float* x, int N, int C) {
    bn_stats_k<<<C, TPB, 0, stream>>>(x, mean, invstd, N, C);
  };
  auto bnbf16 = [&](const float* x, int N, int C, int Cp, int colOff) {
    bnrelu_bf16_k<<<gdim((long long)(N + 1) * C, TPB), TPB, 0, stream>>>(
        x, mean, invstd, xbf, N, C, Cp, colOff);
  };
  auto subconv = [&](const float* Wf, int Cin, int Cinp, int Cout,
                     const int* nb, int N, float* outp) {
    int NT = Cout / 16, KT = Cinp / 32;
    long long pn = (long long)27 * NT * KT * 512;
    pack_w_k<<<gdim(pn, TPB), TPB, 0, stream>>>(Wf, packed, 27, Cin, Cout, Cinp);
    long long waves = (long long)((N + 15) / 16);
    unsigned blocks = gdim(waves * 32, TPB);
    switch (NT) {
      case 2:  subconv_wmma_k<2><<<blocks,  TPB, 0, stream>>>(xbf, nb, packed, outp, N, Cinp); break;
      case 4:  subconv_wmma_k<4><<<blocks,  TPB, 0, stream>>>(xbf, nb, packed, outp, N, Cinp); break;
      case 6:  subconv_wmma_k<6><<<blocks,  TPB, 0, stream>>>(xbf, nb, packed, outp, N, Cinp); break;
      case 8:  subconv_wmma_k<8><<<blocks,  TPB, 0, stream>>>(xbf, nb, packed, outp, N, Cinp); break;
      case 10: subconv_wmma_k<10><<<blocks, TPB, 0, stream>>>(xbf, nb, packed, outp, N, Cinp); break;
      case 12: subconv_wmma_k<12><<<blocks, TPB, 0, stream>>>(xbf, nb, packed, outp, N, Cinp); break;
      case 14: subconv_wmma_k<14><<<blocks, TPB, 0, stream>>>(xbf, nb, packed, outp, N, Cinp); break;
      default: break; // not reachable for this network
    }
  };
  auto linear = [&](const float* X, const float* W, const float* b, float* Y,
                    int N, int Cin, int Cout, int act) {
    linear_k<<<gdim((long long)N * Cout, TPB), TPB, 0, stream>>>(X, W, b, Y, N, Cin, Cout, act);
  };

  // ================= forward =================
  // input conv: feats (N0 x 3) -> actA (N0 x 32)
  convert_pad_k<<<gdim((long long)(N0 + 1) * 32, TPB), TPB, 0, stream>>>(feats, xbf, N0, 3, 32);
  subconv(W_in, 3, 32, 32, nbr[0], N0, actA);

  float* xA = actA;
  float* xB = actB;

  // encoder
  for (int l = 0; l < 6; ++l) {
    int N = Nl[l], C = P[l];
    stats(xA, N, C);
    bnbf16(xA, N, C, C, 0);
    subconv(W_pre[l], C, C, C, nbr[l], N, skip[l]);
    stats(skip[l], N, C);
    bnbf16(skip[l], N, C, C, 0);
    int Nn = Nl[l + 1], Cn = P[l + 1];
    zero_f32_k<<<gdim((long long)Nn * Cn, TPB), TPB, 0, stream>>>(xB, (long long)Nn * Cn);
    downconv_k<<<gdim((long long)N * Cn, TPB), TPB, 0, stream>>>(
        xbf, parent[l], offs[l], W_down[l], xB, N, C, C, Cn);
    float* t = xA; xA = xB; xB = t;
  }

  // deepest level
  {
    int N = Nl[6], C = P[6];
    stats(xA, N, C);
    bnbf16(xA, N, C, C, 0);
    subconv(W_pre[6], C, C, C, nbr[6], N, xB);
    float* t = xA; xA = xB; xB = t;
  }

  // decoder
  for (int l = 5; l >= 0; --l) {
    int Nc = Nl[l + 1], Cc = P[l + 1];
    int Nf = Nl[l],     Cf = P[l];
    stats(xA, Nc, Cc);
    bnbf16(xA, Nc, Cc, Cc, 0);
    deconv_k<<<gdim((long long)Nf * Cf, TPB), TPB, 0, stream>>>(
        xbf, parent[l], offs[l], W_up[l], uBuf, Nf, Cc, Cc, Cf);
    // concat [skip | u] with per-half BN+ReLU into a 2*Cf-wide bf16 buffer
    stats(skip[l], Nf, Cf);
    bnbf16(skip[l], Nf, Cf, 2 * Cf, 0);
    stats(uBuf, Nf, Cf);
    bnbf16(uBuf, Nf, Cf, 2 * Cf, Cf);
    subconv(W_post[l], 2 * Cf, 2 * Cf, Cf, nbr[l], Nf, xB);
    float* t = xA; xA = xB; xB = t;
  }

  // heads: d_out = [y(20) | feature(32) | embedding(32) | off(1) | disp(3)] * N0
  float* out = (float*)d_out;
  float* feature = out + (size_t)20 * N0;
  stats(xA, N0, 32);
  bnrelu_f32_k<<<gdim((long long)N0 * 32, TPB), TPB, 0, stream>>>(xA, mean, invstd, feature, N0, 32);

  linear(feature, Wc, bc, out, N0, 32, 20, 0);
  float* tmp = xB;
  linear(feature, Wfe, bfe, tmp, N0, 32, 32, 0);
  linear(tmp, Wle, ble, out + (size_t)52 * N0, N0, 32, 32, 0);
  linear(feature, Wfr, bfr, tmp, N0, 32, 32, 0);
  linear(tmp, Wlr, blr, out + (size_t)84 * N0, N0, 32, 1, 1);
  linear(feature, Wfd, bfd, tmp, N0, 32, 32, 0);
  linear(tmp, Wld, bld, out + (size_t)85 * N0, N0, 32, 3, 0);
}
